// GINLayer_86225763434891
// MI455X (gfx1250) — compile-verified, matrix operand order checked by reading, not run on hardware
//
#include <hip/hip_runtime.h>

#define N_NODES 100000
#define N_EDGES 600000
#define D 128
#define LN_EPS 1e-5f

typedef __attribute__((ext_vector_type(16))) _Float16 v16h;
typedef __attribute__((ext_vector_type(8)))  _Float16 v8h;
typedef __attribute__((ext_vector_type(8)))  float    v8f;
typedef __attribute__((ext_vector_type(4)))  float    v4f;

// ---------------------------------------------------------------------------
// WMMA fragment helpers (wave32, V_WMMA_F32_16X16X32_F16)
//
// A (16x32, MxK) 16-bit layout: lanes 0-15 hold M=lane, K=kb..kb+7 in halves
// 0..7 and K=16+kb..16+kb+7 in halves 8..15, where kb = (lane>=16)*8.
// B (32x16, KxN) mirrors A with N in place of M (B^T stored like A).
// C/D (16x16 f32): lane 0-15 -> N=lane, M=vgpr; lane 16-31 -> N=lane-16, M=8+vgpr.
// ---------------------------------------------------------------------------

__device__ __forceinline__ v16h hcombine(v8h lo, v8h hi) {
  return __builtin_shufflevector(lo, hi, 0, 1, 2, 3, 4, 5, 6, 7,
                                         8, 9, 10, 11, 12, 13, 14, 15);
}

// A-fragment from an fp32 row-major [N_NODES x D] matrix (convert to f16).
__device__ __forceinline__ v16h load_a_frag_f32(const float* __restrict__ base,
                                                int row0, int k0, int lane) {
  int r = row0 + (lane & 15);
  r = r < N_NODES ? r : N_NODES - 1;            // clamp tail rows (stores guarded)
  int kb = k0 + ((lane >> 4) << 3);
  const float* p = base + (size_t)r * D + kb;
  v4f x0 = *(const v4f*)(p);
  v4f x1 = *(const v4f*)(p + 4);
  v4f x2 = *(const v4f*)(p + 16);
  v4f x3 = *(const v4f*)(p + 20);
  v16h a;
#pragma unroll
  for (int i = 0; i < 4; ++i) {
    a[i]      = (_Float16)x0[i];
    a[4 + i]  = (_Float16)x1[i];
    a[8 + i]  = (_Float16)x2[i];
    a[12 + i] = (_Float16)x3[i];
  }
  return a;
}

// A-fragment from an f16 row-major [N_NODES x D] matrix.
__device__ __forceinline__ v16h load_a_frag_f16(const _Float16* __restrict__ base,
                                                int row0, int k0, int lane) {
  int r = row0 + (lane & 15);
  r = r < N_NODES ? r : N_NODES - 1;
  int kb = k0 + ((lane >> 4) << 3);
  const _Float16* p = base + (size_t)r * D + kb;
  v8h lo = *(const v8h*)(p);
  v8h hi = *(const v8h*)(p + 16);
  return hcombine(lo, hi);
}

// Stage fp32 weight matrix W[k][n] (DxD) into LDS as f16, pre-swizzled so each
// B fragment (n-tile t, k-chunk c, lane) is 16 contiguous halves (2x ds_load_b128).
// Linear LDS index: p = ((t*4 + c)*32 + lane)*16 + h
__device__ __forceinline__ void stage_w_lds(_Float16* lwb, const float* __restrict__ W,
                                            int tid) {
  for (int p = tid; p < D * D; p += 256) {
    int h = p & 15;
    int l = (p >> 4) & 31;
    int c = (p >> 9) & 3;
    int t = p >> 11;
    int kb = (l >> 4) << 3;
    int kk = (h < 8) ? (kb + h) : (16 + kb + (h - 8));
    int k = c * 32 + kk;
    int n = t * 16 + (l & 15);
    lwb[p] = (_Float16)W[k * D + n];
  }
}

// ---------------------------------------------------------------------------
// Kernel 1: m = (1+eps)*x  (eps = 0 -> plain copy, vectorized)
// ---------------------------------------------------------------------------
__global__ void gin_init_m(const float* __restrict__ x, float* __restrict__ m) {
  int i = blockIdx.x * blockDim.x + threadIdx.x;      // exactly N_NODES*D/4 threads
  ((v4f*)m)[i] = ((const v4f*)x)[i];
}

// ---------------------------------------------------------------------------
// Kernel 2: scatter-sum aggregation.  One wave per edge; each lane handles a
// float4 slice of the source row and does 4 hardware f32 global atomics.
// ---------------------------------------------------------------------------
__global__ void gin_scatter(const float* __restrict__ x, const int* __restrict__ ei,
                            float* __restrict__ m) {
  int tid = blockIdx.x * blockDim.x + threadIdx.x;    // exactly N_EDGES*32 threads
  int e = tid >> 5;
  int j = (tid & 31) << 2;
  int src = ei[e];                                     // edge_index[0][e]
  int dst = ei[N_EDGES + e];                           // edge_index[1][e]
  v4f xv = *(const v4f*)(x + (size_t)src * D + j);
  float* mp = m + (size_t)dst * D + j;
#pragma unroll
  for (int i = 0; i < 4; ++i)
    __hip_atomic_fetch_add(&mp[i], xv[i], __ATOMIC_RELAXED, __HIP_MEMORY_SCOPE_AGENT);
}

// ---------------------------------------------------------------------------
// Kernel 3: h1 = relu(m @ W1 + b1), h1 stored as f16.
// 256 threads = 8 waves; each wave owns 16 rows x 128 cols (8 WMMA n-tiles).
// All 8 accumulator tiles kept live so the epilogue can guard once per row
// (8 stores per exec-mask toggle instead of 1).
// ---------------------------------------------------------------------------
__global__ __launch_bounds__(256) void gin_gemm1_relu(const float* __restrict__ m,
                                                      const float* __restrict__ W1,
                                                      const float* __restrict__ b1,
                                                      _Float16* __restrict__ h1) {
  __shared__ _Float16 lwb[D * D] __attribute__((aligned(32)));
  int tid = threadIdx.x;
  stage_w_lds(lwb, W1, tid);
  __syncthreads();

  int lane = tid & 31;
  int wave = tid >> 5;
  int row0 = (blockIdx.x * 8 + wave) * 16;

  v16h a[4];
#pragma unroll
  for (int c = 0; c < 4; ++c) a[c] = load_a_frag_f32(m, row0, c * 32, lane);

  int nlo = lane & 15;
  int rbase = row0 + ((lane >> 4) << 3);

  v8f acc[8];
  float bn[8];
#pragma unroll
  for (int t = 0; t < 8; ++t) {
    acc[t] = (v8f){};
#pragma unroll
    for (int c = 0; c < 4; ++c) {
      v16h b = *(const v16h*)&lwb[((t * 4 + c) * 32 + lane) * 16];
      acc[t] = __builtin_amdgcn_wmma_f32_16x16x32_f16(false, a[c], false, b,
                                                      (short)0, acc[t], false, false);
    }
    bn[t] = b1[t * 16 + nlo];
  }

#pragma unroll
  for (int v = 0; v < 8; ++v) {
    int r = rbase + v;
    if (r < N_NODES) {
      _Float16* hp = h1 + (size_t)r * D;
#pragma unroll
      for (int t = 0; t < 8; ++t) {
        float val = acc[t][v] + bn[t];
        val = val > 0.0f ? val : 0.0f;
        hp[t * 16 + nlo] = (_Float16)val;
      }
    }
  }
}

// ---------------------------------------------------------------------------
// Kernel 4: out = LayerNorm(h1 @ W2 + b2) * gamma + beta, fused.
// Keeps all 8 f32 accumulator tiles live (64 VGPRs); row statistics reduce
// across the 16 lanes of each half-wave with 4 xor-shuffles.
// ---------------------------------------------------------------------------
__global__ __launch_bounds__(256) void gin_gemm2_ln(const _Float16* __restrict__ h1,
                                                    const float* __restrict__ W2,
                                                    const float* __restrict__ b2,
                                                    const float* __restrict__ gamma,
                                                    const float* __restrict__ beta,
                                                    float* __restrict__ out) {
  __shared__ _Float16 lwb[D * D] __attribute__((aligned(32)));
  int tid = threadIdx.x;
  stage_w_lds(lwb, W2, tid);
  __syncthreads();

  int lane = tid & 31;
  int wave = tid >> 5;
  int row0 = (blockIdx.x * 8 + wave) * 16;

  v16h a[4];
#pragma unroll
  for (int c = 0; c < 4; ++c) a[c] = load_a_frag_f16(h1, row0, c * 32, lane);

  int nlo = lane & 15;
  int rbase = row0 + ((lane >> 4) << 3);

  v8f acc[8];
#pragma unroll
  for (int t = 0; t < 8; ++t) {
    acc[t] = (v8f){};
#pragma unroll
    for (int c = 0; c < 4; ++c) {
      v16h b = *(const v16h*)&lwb[((t * 4 + c) * 32 + lane) * 16];
      acc[t] = __builtin_amdgcn_wmma_f32_16x16x32_f16(false, a[c], false, b,
                                                      (short)0, acc[t], false, false);
    }
    float bnv = b2[t * 16 + nlo];
#pragma unroll
    for (int v = 0; v < 8; ++v) acc[t][v] += bnv;
  }

  float g[8], be[8];
#pragma unroll
  for (int t = 0; t < 8; ++t) {
    g[t]  = gamma[t * 16 + nlo];
    be[t] = beta[t * 16 + nlo];
  }

#pragma unroll
  for (int v = 0; v < 8; ++v) {
    float s = 0.0f, ss = 0.0f;
#pragma unroll
    for (int t = 0; t < 8; ++t) {
      float xv = acc[t][v];
      s += xv;
      ss += xv * xv;
    }
#pragma unroll
    for (int msk = 1; msk < 16; msk <<= 1) {   // masks 1,2,4,8: stays in half-wave
      s  += __shfl_xor(s,  msk, 32);
      ss += __shfl_xor(ss, msk, 32);
    }
    float mu  = s * (1.0f / D);
    float var = ss * (1.0f / D) - mu * mu;
    float rs  = rsqrtf(var + LN_EPS);
    int r = rbase + v;
    if (r < N_NODES) {
      float* op = out + (size_t)r * D;
#pragma unroll
      for (int t = 0; t < 8; ++t)
        op[t * 16 + nlo] = (acc[t][v] - mu) * rs * g[t] + be[t];
    }
  }
}

// ---------------------------------------------------------------------------
extern "C" void kernel_launch(void* const* d_in, const int* in_sizes, int n_in,
                              void* d_out, int out_size, void* d_ws, size_t ws_size,
                              hipStream_t stream) {
  (void)in_sizes; (void)n_in; (void)out_size; (void)ws_size;
  const float* x     = (const float*)d_in[0];
  const float* W1    = (const float*)d_in[1];
  const float* b1    = (const float*)d_in[2];
  const float* W2    = (const float*)d_in[3];
  const float* b2    = (const float*)d_in[4];
  const float* gamma = (const float*)d_in[5];
  const float* beta  = (const float*)d_in[6];
  const int*   ei    = (const int*)d_in[7];
  // d_in[8] = batch (unused: node-mode LayerNorm)

  float*     m  = (float*)d_ws;                                       // 51.2 MB f32
  _Float16*  h1 = (_Float16*)((char*)d_ws + (size_t)N_NODES * D * 4); // 25.6 MB f16
  float*     out = (float*)d_out;

  gin_init_m<<<(N_NODES * D / 4) / 256, 256, 0, stream>>>(x, m);
  gin_scatter<<<(N_EDGES * 32) / 256, 256, 0, stream>>>(x, ei, m);

  int gblocks = (N_NODES + 127) / 128;   // 128 rows per 8-wave block
  gin_gemm1_relu<<<gblocks, 256, 0, stream>>>(m, W1, b1, h1);
  gin_gemm2_ln<<<gblocks, 256, 0, stream>>>(h1, W2, b2, gamma, beta, out);
}
